// AttentionAggregatorVectorized_72138270703926
// MI455X (gfx1250) — matrix operand order repaired and checked by reading, not compiled, not run on hardware
//
#include <hip/hip_runtime.h>
#include <hip/hip_bf16.h>

// ---------------------------------------------------------------------------
// GAT-style attention aggregator for MI455X (gfx1250, wave32, WMMA).
// B=4096, N=20, G=4, D_IN=128, D_OUT=64, H=8.
// Strategy: one workgroup (512 thr = 16 wave32) per batch element b.
//   - X rows [96 x 128] (80 neigh + 4 self + 12 zero-pad) staged in LDS, bf16.
//   - W_heads pre-transposed to bf16 Wt[c=h*64+e][k] in d_ws (prep kernel);
//     L2-resident (131 KB), B-fragments loaded straight from global.
//   - Wave w: head h=w/2, column tiles {2*(w&1), 2*(w&1)+1}.
//     GEMM: 6 row-tiles x 2 col-tiles x 4 K-steps = 48 v_wmma_f32_16x16x32_bf16.
//   - Attention epilogue fully in registers + small LDS exchange (no atomics).
// ---------------------------------------------------------------------------

typedef __attribute__((ext_vector_type(16))) __bf16 v16bf;
typedef __attribute__((ext_vector_type(8)))  float  v8f;

union Frag {
    unsigned int u[8];
    v16bf        v;
};

#define BATCH   4096
#define NNEIGH  20
#define NG      4
#define DIN     128
#define DOUT    64
#define NHEAD   8
#define LRELU_A 0.2f

#define XROWS   96          // 80 neigh rows + 4 self rows + 12 pad rows
#define XSTRIDE 134         // ushorts per row; 67 dwords (odd) -> no LDS bank conflicts

__device__ __forceinline__ unsigned short f32_to_bf16(float f) {
    unsigned int u = __float_as_uint(f);
    u += 0x7FFFu + ((u >> 16) & 1u);     // round-to-nearest-even
    return (unsigned short)(u >> 16);
}

// --- prep: W_heads [H,128,64] f32 -> Wt bf16, layout Wt[c][k], c=h*64+e -----
__global__ void gat_prep_w(const float* __restrict__ W, unsigned short* __restrict__ Wt) {
    int i = blockIdx.x * blockDim.x + threadIdx.x;      // 0 .. 8*64*128-1
    if (i >= NHEAD * DOUT * DIN) return;
    int c = i >> 7;            // column (h*64+e)
    int k = i & 127;           // K index
    int h = c >> 6;
    int e = c & 63;
    Wt[i] = f32_to_bf16(W[(h * DIN + k) * DOUT + e]);
}

// --- main kernel ------------------------------------------------------------
__global__ __launch_bounds__(512)
void gat_main(const float* __restrict__ self_vecs,    // [B,G,128]
              const float* __restrict__ neigh_vecs,   // [B,N,G,128]
              const float* __restrict__ a_self,       // [H,64]
              const float* __restrict__ a_neigh,      // [H,64]
              const float* __restrict__ W_out,        // [512,64]
              const unsigned short* __restrict__ Wt,  // [512,128] bf16 (transposed W)
              float* __restrict__ out)                // [B,G,64]
{
    __shared__ unsigned short Xs[XROWS * XSTRIDE];    // 25728 B, bf16 X rows
    __shared__ float f_lds[2 * NHEAD * 96];           // partial f, slot = wave parity
    __shared__ float attend_lds[NHEAD * NNEIGH * NG]; // softmax weights
    __shared__ float hbuf[NG * NHEAD * DOUT];         // [4,512] pre-FC activations

    const int b    = blockIdx.x;
    const int tid  = threadIdx.x;
    const int w    = tid >> 5;          // wave 0..15
    const int lane = tid & 31;
    const int lm   = lane & 15;
    const int lhalf = lane >> 4;
    const int h      = w >> 1;          // head 0..7
    const int ctbase = (w & 1) << 1;    // first of this wave's 2 column tiles

    // ---- Phase A: stage X (fp32 -> bf16) into LDS ----------------------------
    {
        unsigned int* Xd = (unsigned int*)Xs;
        const float2* nsrc = (const float2*)(neigh_vecs) + (size_t)b * (NNEIGH * NG * DIN / 2);
        const float2* ssrc = (const float2*)(self_vecs)  + (size_t)b * (NG * DIN / 2);
        // 5120 neigh pairs + 256 self pairs + 768 zero pairs
        for (int idx = tid; idx < 5120 + 256 + 768; idx += 512) {
            int r, kp;
            unsigned int packed;
            if (idx < 5120) {
                r = idx >> 6; kp = idx & 63;
                float2 v = nsrc[idx];
                packed = (unsigned int)f32_to_bf16(v.x) |
                         ((unsigned int)f32_to_bf16(v.y) << 16);
            } else if (idx < 5376) {
                int i2 = idx - 5120;
                r = 80 + (i2 >> 6); kp = i2 & 63;
                float2 v = ssrc[i2];
                packed = (unsigned int)f32_to_bf16(v.x) |
                         ((unsigned int)f32_to_bf16(v.y) << 16);
            } else {
                int i2 = idx - 5376;
                r = 84 + (i2 >> 6); kp = i2 & 63;
                packed = 0u;
            }
            Xd[r * (XSTRIDE / 2) + kp] = packed;     // XSTRIDE/2 = 67 dwords/row
        }
    }
    __syncthreads();

    // ---- per-lane attention weight scalars (column e = ct*16 + lm) ----------
    float anv[2], asv[2];
#pragma unroll
    for (int ci = 0; ci < 2; ++ci) {
        int e = (ctbase + ci) * 16 + lm;
        anv[ci] = a_neigh[h * DOUT + e];
        asv[ci] = a_self [h * DOUT + e];
    }

    // ---- Phase B: GEMM  T = X @ W   (bf16 WMMA, f32 accumulate) -------------
    v8f acc[6][2];
#pragma unroll
    for (int rt = 0; rt < 6; ++rt)
#pragma unroll
        for (int ci = 0; ci < 2; ++ci)
            acc[rt][ci] = {};

#pragma unroll
    for (int ks = 0; ks < 4; ++ks) {
        // B fragments (K=32 x N=16): lane = column, VGPR v holds K0=16*(v/4)+2*(v%4)+8*half
        Frag Bf[2];
#pragma unroll
        for (int ci = 0; ci < 2; ++ci) {
            int c = h * DOUT + (ctbase + ci) * 16 + lm;
#pragma unroll
            for (int v = 0; v < 8; ++v) {
                int K0 = ((v >> 2) << 4) + ((v & 3) << 1) + (lhalf << 3);
                Bf[ci].u[v] = *(const unsigned int*)(Wt + c * DIN + ks * 32 + K0);
            }
        }
#pragma unroll
        for (int rt = 0; rt < 6; ++rt) {
            // A fragment (M=16 x K=32): lane = row, VGPR v: K0=16*(v/4)+2*(v%4)+8*half
            Frag Af;
#pragma unroll
            for (int v = 0; v < 8; ++v) {
                int K0 = ((v >> 2) << 4) + ((v & 3) << 1) + (lhalf << 3);
                Af.u[v] = *(const unsigned int*)&Xs[(rt * 16 + lm) * XSTRIDE + ks * 32 + K0];
            }
#pragma unroll
            for (int ci = 0; ci < 2; ++ci) {
                acc[rt][ci] = __builtin_amdgcn_wmma_f32_16x16x32_bf16(
                    false, Af.v, false, Bf[ci].v, (short)0, acc[rt][ci], false, false);
            }
        }
    }

    // ---- f = T . a  (partial over this wave's 32 columns) -------------------
    // C layout: VGPR j -> row = rt*16 + j + 8*half, column e = ct*16 + lm.
    {
        const int slot = (w & 1) * (NHEAD * 96);
#pragma unroll
        for (int rt = 0; rt < 6; ++rt) {
            float aw0 = (rt == 5) ? asv[0] : anv[0];   // rows >= 80 are self rows
            float aw1 = (rt == 5) ? asv[1] : anv[1];
#pragma unroll
            for (int j = 0; j < 8; ++j) {
                float fp = acc[rt][0][j] * aw0 + acc[rt][1][j] * aw1;
                fp += __shfl_xor(fp, 1);
                fp += __shfl_xor(fp, 2);
                fp += __shfl_xor(fp, 4);
                fp += __shfl_xor(fp, 8);               // sum within 16-lane half
                if (lm == 0)
                    f_lds[slot + h * 96 + rt * 16 + j + (lhalf << 3)] = fp;
            }
        }
    }
    __syncthreads();

    // ---- softmax over G (axis=2), with leaky-relu on logits ----------------
    if (((w & 1) == 0) && lane < NNEIGH) {
        int n = lane;
        float e4[4], den = 0.f;
#pragma unroll
        for (int g = 0; g < 4; ++g) {
            float fs = f_lds[h * 96 + 80 + g]    + f_lds[NHEAD * 96 + h * 96 + 80 + g];
            float fn = f_lds[h * 96 + n * 4 + g] + f_lds[NHEAD * 96 + h * 96 + n * 4 + g];
            float l = fs + fn;
            l = (l > 0.f) ? l : LRELU_A * l;
            e4[g] = __expf(l);
            den += e4[g];
        }
        float inv = 1.f / den;
#pragma unroll
        for (int g = 0; g < 4; ++g)
            attend_lds[(h * NNEIGH + n) * 4 + g] = e4[g] * inv;
    }
    __syncthreads();

    // ---- weighted aggregation over N, leaky-relu, write hbuf ----------------
    // row = rt*16 + jj + 8*half with jj in {g, g+4}  =>  n = 4*rt + 2*half + (jj>=4)
#pragma unroll
    for (int ci = 0; ci < 2; ++ci) {
#pragma unroll
        for (int g = 0; g < 4; ++g) {
            float s = 0.f;
#pragma unroll
            for (int rt = 0; rt < 5; ++rt) {           // rt=5 holds self rows: excluded
                int n0 = 4 * rt + (lhalf << 1);
                s += attend_lds[(h * NNEIGH + n0)     * 4 + g] * acc[rt][ci][g];
                s += attend_lds[(h * NNEIGH + n0 + 1) * 4 + g] * acc[rt][ci][g + 4];
            }
            s += __shfl_xor(s, 16);                    // combine lane halves
            float o = (s > 0.f) ? s : LRELU_A * s;
            if (lane < 16)
                hbuf[g * (NHEAD * DOUT) + h * DOUT + (ctbase + ci) * 16 + lane] = o;
        }
    }
    __syncthreads();

    // ---- final FC: [4,512] @ [512,64] + relu --------------------------------
    if (tid < NG * DOUT) {
        int g  = tid >> 6;
        int eo = tid & 63;
        float s = 0.f;
        const float* hrow = &hbuf[g * (NHEAD * DOUT)];
#pragma unroll 8
        for (int k = 0; k < NHEAD * DOUT; ++k)
            s += hrow[k] * W_out[k * DOUT + eo];
        out[((b << 2) + g) * DOUT + eo] = (s > 0.f) ? s : 0.f;
    }
}

extern "C" void kernel_launch(void* const* d_in, const int* in_sizes, int n_in,
                              void* d_out, int out_size, void* d_ws, size_t ws_size,
                              hipStream_t stream) {
    const float* self_vecs = (const float*)d_in[0];
    const float* neigh_vecs = (const float*)d_in[1];
    const float* W_heads   = (const float*)d_in[2];
    const float* a_self    = (const float*)d_in[3];
    const float* a_neigh   = (const float*)d_in[4];
    const float* W_out     = (const float*)d_in[5];

    unsigned short* Wt = (unsigned short*)d_ws;       // 512*128 bf16 = 131072 B

    gat_prep_w<<<(NHEAD * DOUT * DIN + 255) / 256, 256, 0, stream>>>(W_heads, Wt);
    gat_main<<<BATCH, 512, 0, stream>>>(self_vecs, neigh_vecs, a_self, a_neigh,
                                        W_out, Wt, (float*)d_out);
}
